// XiRNNBase_45500883534367
// MI455X (gfx1250) — matrix operand order repaired and checked by reading, not compiled, not run on hardware
//
#include <hip/hip_runtime.h>
#include <hip/hip_bf16.h>

typedef __attribute__((ext_vector_type(16))) __bf16 v16bf;
typedef __attribute__((ext_vector_type(8)))  float  v8f;

#define BB 16
#define TT 1024
#define II 512
#define HH 1024
#define G4 4096
#define BTH (BB*TT*HH)

// Packed element counts (bf16)
#define N_PK_WHH  (256*32*512)   // 256 n-tiles, 32 k-tiles, 32 lanes * 16 vals
#define N_PK_WREC (128*32*512)
#define N_PK_WIH  (256*16*512)
#define N_PK_X    (1024*16*512)  // 1024 m-tiles, 16 k-tiles
#define N_PK_ST   16384          // one packed 16x1024 state matrix (A layout)

// Workspace byte offsets
#define OFF_WHH   0
#define OFF_WREC  (OFF_WHH  + (size_t)N_PK_WHH*2)
#define OFF_WIH   (OFF_WREC + (size_t)N_PK_WREC*2)
#define OFF_X     (OFF_WIH  + (size_t)N_PK_WIH*2)
#define OFF_HPACK (OFF_X    + (size_t)N_PK_X*2)        // 2 ping-pong h buffers
#define OFF_CPACK (OFF_HPACK + (size_t)2*N_PK_ST*2)    // 2 ping-pong c buffers
#define OFF_PREX  (OFF_CPACK + (size_t)2*N_PK_ST*2)

// ---------------------------------------------------------------------------
// Pack weights into WMMA B-fragment layout:
//   B frag (32x16, K x N): lane l, element e -> N = l&15, K = e + 8*((e>=8) + (l>=16))
// ---------------------------------------------------------------------------
__global__ __launch_bounds__(256) void pack_weights_kernel(
    const float* __restrict__ whh, const float* __restrict__ wrec,
    const float* __restrict__ wih,
    __bf16* __restrict__ p_whh, __bf16* __restrict__ p_wrec,
    __bf16* __restrict__ p_wih)
{
    const int total = N_PK_WHH + N_PK_WREC + N_PK_WIH;
    for (int i = blockIdx.x * 256 + threadIdx.x; i < total; i += gridDim.x * 256) {
        int j = i;
        const float* src; __bf16* dst; int kt_bits, Kdim;
        if (j < N_PK_WHH)                      { src = whh;  dst = p_whh;  kt_bits = 5; Kdim = 1024; }
        else if (j < N_PK_WHH + N_PK_WREC)     { j -= N_PK_WHH; src = wrec; dst = p_wrec; kt_bits = 5; Kdim = 1024; }
        else { j -= (N_PK_WHH + N_PK_WREC);      src = wih;  dst = p_wih;  kt_bits = 4; Kdim = 512; }
        int e  = j & 15;
        int l  = (j >> 4) & 31;
        int kt = (j >> 9) & ((1 << kt_bits) - 1);
        int nt = j >> (9 + kt_bits);
        int n  = nt * 16 + (l & 15);
        int k  = kt * 32 + e + 8 * ((e >= 8) + (l >= 16));
        dst[j] = (__bf16)src[(long)n * Kdim + k];
    }
}

// ---------------------------------------------------------------------------
// Pack x into WMMA A-fragment layout:
//   A frag (16x32, M x K): lane l, element e ->
//     row = l&15, k = e + 8*((e>=8) + (l>=16))
// ---------------------------------------------------------------------------
__global__ __launch_bounds__(256) void pack_x_kernel(
    const float* __restrict__ x, __bf16* __restrict__ px)
{
    for (int i = blockIdx.x * 256 + threadIdx.x; i < N_PK_X; i += gridDim.x * 256) {
        int e  = i & 15;
        int l  = (i >> 4) & 31;
        int kt = (i >> 9) & 15;
        int mt = i >> 13;
        int m  = mt * 16 + (l & 15);
        int k  = kt * 32 + e + 8 * ((e >= 8) + (l >= 16));
        px[i] = (__bf16)x[(long)m * II + k];
    }
}

// ---------------------------------------------------------------------------
// pre_x GEMM: [16384 x 512] x [512 x 4096] + bias -> preX stored [T][B][4H]
// 8 waves/block; each wave: 1 m-tile x 4 n-tiles, full K.
// Grid: (1024 m-tiles, 8)  ->  8*4=32 n-tiles per block, 8 blocks cover 256.
// All inner-loop loads use one base VGPR address + constant immediate offsets.
// ---------------------------------------------------------------------------
__global__ __launch_bounds__(256) void prex_gemm_kernel(
    const __bf16* __restrict__ px, const __bf16* __restrict__ pw,
    const float* __restrict__ bias, float* __restrict__ preX)
{
    const int tid  = threadIdx.x;
    const int wave = tid >> 5;
    const int lane = tid & 31;
    const int mt   = blockIdx.x;
    const int ntb  = blockIdx.y * 32 + wave * 4;

    // Per-wave base pointers; every load below is base + compile-time offset.
    const __bf16* aB = px + ((size_t)(mt * 16) * 32 + lane) * 16;
    const __bf16* bB = pw + ((size_t)(ntb * 16) * 32 + lane) * 16;

    v8f acc[4] = {v8f{}, v8f{}, v8f{}, v8f{}};
#pragma unroll
    for (int kt = 0; kt < 16; ++kt) {
        v16bf a = *(const v16bf*)(aB + kt * 512);
#pragma unroll
        for (int j = 0; j < 4; ++j) {
            v16bf b = *(const v16bf*)(bB + j * 8192 + kt * 512);
            acc[j] = __builtin_amdgcn_wmma_f32_16x16x32_bf16(
                false, a, false, b, (short)0, acc[j], false, false);
        }
    }
    const int Mh = 8 * (lane >> 4);
    const int N  = lane & 15;
#pragma unroll
    for (int j = 0; j < 4; ++j) {
        const int n = (ntb + j) * 16 + N;
        const float bv = bias[n];
#pragma unroll
        for (int i = 0; i < 8; ++i) {
            int m = mt * 16 + i + Mh;
            int b = m >> 10;          // T = 1024 rows per batch
            int t = m & 1023;
            preX[(long)(t * 16 + b) * G4 + n] = acc[j][i] + bv;
        }
    }
}

// ---------------------------------------------------------------------------
// One recurrent timestep. 64 blocks x 384 threads (12 waves).
//   Waves 0..7 : Whh GEMM, wave = gate*2 + k_half  (A = packed h_{t-1})
//   Waves 8..11: Wrec GEMM, wave-8 = peep*2 + k_half (A = packed c_{t-1})
// A fragments come pre-packed in bf16 from the previous step's output
// (no LDS staging, no scatter: pure global_load_b128 -> v_wmma).
// Then 256 threads do gate math; write f32 h/c to d_out AND bf16
// A-fragment-packed h/c for the next step (ping-pong buffers).
// ---------------------------------------------------------------------------
__global__ __launch_bounds__(384) void lstm_step_kernel(
    const __bf16* __restrict__ p_whh, const __bf16* __restrict__ p_wrec,
    const float* __restrict__ preX,
    const __bf16* __restrict__ hpk_prev, const __bf16* __restrict__ cpk_prev,
    __bf16* __restrict__ hpk_next, __bf16* __restrict__ cpk_next,
    float* __restrict__ out, int t)
{
    __shared__ float Ct[8][16][16];    // gate partials (4 gates x 2 K-halves)
    __shared__ float Ct2[4][16][16];   // peephole partials (2 peeps x 2 K-halves)

    const int tid  = threadIdx.x;
    const int wave = tid >> 5;
    const int lane = tid & 31;
    const int wgj  = blockIdx.x;       // 0..63 hidden-column tile

    const float* cs = out + BTH;       // [B][T][H]

    if (wave < 8) {
        // ---- Whh: pre[:, gate*H + wgj*16 .. +15] ----
        const int gate  = wave >> 1;
        const int kh    = wave & 1;
        const int ntile = gate * 64 + wgj;
        const __bf16* aB = hpk_prev + ((size_t)(kh * 16) * 32 + lane) * 16;
        const __bf16* bB = p_whh +
            (((size_t)ntile * 32 + kh * 16) * 32 + lane) * 16;
        v8f acc = {};
        if (t > 0) {
#pragma unroll
            for (int kk = 0; kk < 16; ++kk) {
                v16bf a = *(const v16bf*)(aB + kk * 512);
                v16bf b = *(const v16bf*)(bB + kk * 512);
                acc = __builtin_amdgcn_wmma_f32_16x16x32_bf16(
                    false, a, false, b, (short)0, acc, false, false);
            }
        }
        const int Mh = 8 * (lane >> 4);
        const int N  = lane & 15;
#pragma unroll
        for (int i = 0; i < 8; ++i) Ct[wave][i + Mh][N] = acc[i];
    } else {
        // ---- Wrec peephole: pre_c[:, peep*H + wgj*16 .. +15] ----
        const int w2    = wave - 8;
        const int pg    = w2 >> 1;
        const int kh    = w2 & 1;
        const int ntile = pg * 64 + wgj;
        const __bf16* aB = cpk_prev + ((size_t)(kh * 16) * 32 + lane) * 16;
        const __bf16* bB = p_wrec +
            (((size_t)ntile * 32 + kh * 16) * 32 + lane) * 16;
        v8f acc = {};
        if (t > 0) {
#pragma unroll
            for (int kk = 0; kk < 16; ++kk) {
                v16bf a = *(const v16bf*)(aB + kk * 512);
                v16bf b = *(const v16bf*)(bB + kk * 512);
                acc = __builtin_amdgcn_wmma_f32_16x16x32_bf16(
                    false, a, false, b, (short)0, acc, false, false);
            }
        }
        const int Mh = 8 * (lane >> 4);
        const int N  = lane & 15;
#pragma unroll
        for (int i = 0; i < 8; ++i) Ct2[w2][i + Mh][N] = acc[i];
    }
    __syncthreads();

    // ---- Gate math: one thread per (batch, hidden-col) ----
    if (tid < 256) {
        const int b  = tid >> 4;
        const int j  = tid & 15;
        const int jg = wgj * 16 + j;
        const long prow = (long)(t * 16 + b) * G4;

        float preI = Ct[0][b][j] + Ct[1][b][j] + Ct2[0][b][j] + Ct2[1][b][j] + preX[prow + jg];
        float preF = Ct[2][b][j] + Ct[3][b][j] + Ct2[2][b][j] + Ct2[3][b][j] + preX[prow + HH + jg];
        float preG = Ct[4][b][j] + Ct[5][b][j] + preX[prow + 2 * HH + jg];
        float preO = Ct[6][b][j] + Ct[7][b][j] + preX[prow + 3 * HH + jg];

        float cold = (t > 0) ? cs[(long)b * TT * HH + (long)(t - 1) * HH + jg] : 0.0f;

        float ig = 1.0f / (1.0f + __expf(-preI));
        float fg = 1.0f / (1.0f + __expf(-preF));
        float gg = tanhf(preG);
        float cn = fg * cold + ig * gg;
        float og = 1.0f / (1.0f + __expf(-(preO + cn)));
        float hn = og * tanhf(cn);

        // f32 results (also the recurrent f32 c state)
        out[(long)b * TT * HH + (long)t * HH + jg]       = hn;
        out[BTH + (long)b * TT * HH + (long)t * HH + jg] = cn;

        // bf16 state pre-packed in A-fragment layout for step t+1:
        //   row = b, k = jg;  kl = k&31:
        //   lhalf = (kl>>3)&1, e = (kl&7) + 8*(kl>=16), l = row + 16*lhalf
        {
            const int kt    = jg >> 5;
            const int kl    = jg & 31;
            const int lhalf = (kl >> 3) & 1;
            const int e     = (kl & 7) + 8 * (kl >= 16);
            const int l     = b + 16 * lhalf;
            const int idx   = (kt * 32 + l) * 16 + e;
            hpk_next[idx] = (__bf16)hn;
            cpk_next[idx] = (__bf16)cn;
        }
    }
}

// ---------------------------------------------------------------------------
extern "C" void kernel_launch(void* const* d_in, const int* in_sizes, int n_in,
                              void* d_out, int out_size, void* d_ws, size_t ws_size,
                              hipStream_t stream) {
    const float* x    = (const float*)d_in[0];   // [B,T,I]
    const float* wih  = (const float*)d_in[1];   // [4H,I]
    const float* whh  = (const float*)d_in[2];   // [4H,H]
    const float* wrec = (const float*)d_in[3];   // [2H,H]
    const float* bias = (const float*)d_in[4];   // [4H]
    float* out = (float*)d_out;                  // hs [B,T,H] ++ cs [B,T,H]

    char* ws = (char*)d_ws;
    __bf16* p_whh  = (__bf16*)(ws + OFF_WHH);
    __bf16* p_wrec = (__bf16*)(ws + OFF_WREC);
    __bf16* p_wih  = (__bf16*)(ws + OFF_WIH);
    __bf16* p_x    = (__bf16*)(ws + OFF_X);
    __bf16* hpk[2] = { (__bf16*)(ws + OFF_HPACK),
                       (__bf16*)(ws + OFF_HPACK) + N_PK_ST };
    __bf16* cpk[2] = { (__bf16*)(ws + OFF_CPACK),
                       (__bf16*)(ws + OFF_CPACK) + N_PK_ST };
    float*  preX   = (float*)(ws + OFF_PREX);    // [T][B][4H]

    pack_weights_kernel<<<4096, 256, 0, stream>>>(whh, wrec, wih, p_whh, p_wrec, p_wih);
    pack_x_kernel<<<4096, 256, 0, stream>>>(x, p_x);

    dim3 gprex(1024, 8);
    prex_gemm_kernel<<<gprex, 256, 0, stream>>>(p_x, p_wih, bias, preX);

    for (int t = 0; t < TT; ++t) {
        const int cur = t & 1, prev = cur ^ 1;
        lstm_step_kernel<<<64, 384, 0, stream>>>(
            p_whh, p_wrec, preX,
            hpk[prev], cpk[prev], hpk[cur], cpk[cur],
            out, t);
    }
}
